// SingleScaleGCN_14070312862325
// MI455X (gfx1250) — matrix operand order repaired and checked by reading, not compiled, not run on hardware
//
#include <hip/hip_runtime.h>

#define NNODES 10000
#define NEDGES 160000
#define EPSBN  1e-5f
#define SLOPE  0.01f

typedef __attribute__((ext_vector_type(16))) __bf16 bf16x16;
typedef __attribute__((ext_vector_type(8)))  float  v8f;

// ---- Tensor Data Mover availability / arity detection (compile-time) ----
#if defined(__HIP_DEVICE_COMPILE__) && __has_builtin(__builtin_amdgcn_tensor_load_to_lds)
#define USE_TDM 1
#else
#define USE_TDM 0
#endif
#if __has_include(<hip/amd_detail/amd_gfx1250_TDM.h>)
#define TDM_SIX_ARGS 1
#else
#define TDM_SIX_ARGS 0
#endif

static __device__ __forceinline__ unsigned short f32_to_bf16(float f) {
  unsigned int u = __float_as_uint(f);
  u += 0x7FFFu + ((u >> 16) & 1u);           // round-to-nearest-even
  return (unsigned short)(u >> 16);
}

#if USE_TDM
typedef __attribute__((ext_vector_type(4))) unsigned int u32x4;
typedef __attribute__((ext_vector_type(8))) int          i32x8;
typedef __attribute__((ext_vector_type(4))) int          i32x4;

// DMA a 2D tile (rows x row_elems of 2-byte elements, row stride = stride_elems)
// from global memory into LDS at byte offset lds_off.  D# per CDNA5 ISA §8.3/§8.4.
static __device__ __forceinline__ void tdm_load_tile_2d(
    const void* gptr, unsigned lds_off, unsigned rows,
    unsigned row_elems, unsigned stride_elems) {
  unsigned long long ga = (unsigned long long)gptr;
  u32x4 g0;
  g0[0] = 1u;                                               // count=1 (valid), user mode
  g0[1] = lds_off;                                          // lds_addr (bytes)
  g0[2] = (unsigned)(ga & 0xFFFFFFFFull);                   // global_addr[31:0]
  g0[3] = (unsigned)((ga >> 32) & 0x01FFFFFFull)            // global_addr[56:32]
        | (2u << 30);                                       // type = 2 ("image")
  i32x8 g1;
  g1[0] = (int)(1u << 16);                                  // data_size=1 (2 bytes), no mask
  g1[1] = (int)((row_elems & 0xFFFFu) << 16);               // tensor_dim0[15:0] @ bit48
  g1[2] = (int)(((row_elems >> 16) & 0xFFFFu)               // tensor_dim0[31:16]
        | ((rows & 0xFFFFu) << 16));                        // tensor_dim1[15:0]
  g1[3] = (int)(((rows >> 16) & 0xFFFFu)                    // tensor_dim1[31:16]
        | ((row_elems & 0xFFFFu) << 16));                   // tile_dim0
  g1[4] = (int)(rows & 0xFFFFu);                            // tile_dim1 (tile_dim2=0)
  g1[5] = (int)stride_elems;                                // tensor_dim0_stride[31:0]
  g1[6] = 0;                                                // stride hi / dim1_stride lo
  g1[7] = 0;
  i32x4 z4 = {0, 0, 0, 0};
#if TDM_SIX_ARGS
  i32x8 z8 = {0, 0, 0, 0, 0, 0, 0, 0};
  __builtin_amdgcn_tensor_load_to_lds(g0, g1, z4, z4, z8, 0);
#else
  __builtin_amdgcn_tensor_load_to_lds(g0, g1, z4, z4, 0);
#endif
}
#endif  // USE_TDM

// ---------------- elementwise / graph kernels ----------------

__global__ void k_zero(float* p, int n) {
  int i = blockIdx.x * blockDim.x + threadIdx.x;
  if (i < n) p[i] = 0.0f;
}

__global__ void k_count_deg(const int* __restrict__ dst, float* deg, int e) {
  int i = blockIdx.x * blockDim.x + threadIdx.x;
  if (i < e) atomicAdd(&deg[dst[i]], 1.0f);
}

__global__ void k_dinv(const float* __restrict__ deg, float* dinv, int n) {
  int i = blockIdx.x * blockDim.x + threadIdx.x;
  if (i < n) dinv[i] = rsqrtf(deg[i] + 1.0f);   // +1 self loop
}

__global__ void k_minmax(const float* __restrict__ z1, float* mm, int n) {
  __shared__ float smn[3][256];
  __shared__ float smx[3][256];
  float mn[3] = { 1e30f, 1e30f, 1e30f };
  float mx[3] = { -1e30f, -1e30f, -1e30f };
  for (int i = threadIdx.x; i < n; i += 256) {
    #pragma unroll
    for (int c = 0; c < 3; ++c) {
      float v = z1[i * 3 + c];
      mn[c] = fminf(mn[c], v);
      mx[c] = fmaxf(mx[c], v);
    }
  }
  #pragma unroll
  for (int c = 0; c < 3; ++c) { smn[c][threadIdx.x] = mn[c]; smx[c][threadIdx.x] = mx[c]; }
  __syncthreads();
  for (int t = 128; t > 0; t >>= 1) {
    if ((int)threadIdx.x < t) {
      #pragma unroll
      for (int c = 0; c < 3; ++c) {
        smn[c][threadIdx.x] = fminf(smn[c][threadIdx.x], smn[c][threadIdx.x + t]);
        smx[c][threadIdx.x] = fmaxf(smx[c][threadIdx.x], smx[c][threadIdx.x + t]);
      }
    }
    __syncthreads();
  }
  if (threadIdx.x == 0) {
    float sc = 0.0f;
    #pragma unroll
    for (int c = 0; c < 3; ++c) {
      mm[c] = (smn[c][0] + smx[c][0]) * 0.5f;       // center
      sc = fmaxf(sc, smx[c][0] - smn[c][0]);        // max extent
    }
    mm[3] = sc;
  }
}

__global__ void k_build_input(const float* __restrict__ z1, const float* __restrict__ dm,
                              const float* __restrict__ mm, float* __restrict__ x, int n) {
  int i = blockIdx.x * blockDim.x + threadIdx.x;
  if (i >= n) return;
  float d = dm[i];
  float inv = 1.0f / mm[3];
  #pragma unroll
  for (int c = 0; c < 3; ++c) x[i * 4 + c] = d * ((z1[i * 3 + c] - mm[c]) * inv);
  x[i * 4 + 3] = d;
}

__global__ void k_cvt_bf16(const float* __restrict__ in, unsigned short* __restrict__ out,
                           int rows, int K, int Kp) {
  int idx = blockIdx.x * blockDim.x + threadIdx.x;
  if (idx >= rows * Kp) return;
  int r = idx / Kp, k = idx - r * Kp;
  out[idx] = (k < K) ? f32_to_bf16(in[(size_t)r * K + k]) : (unsigned short)0;
}

// W [K, Nout] row-major -> Wt bf16 [Nout, Kp] (transposed, K zero-padded)
__global__ void k_cvt_wt(const float* __restrict__ W, unsigned short* __restrict__ out,
                         int K, int Nout, int Kp) {
  int idx = blockIdx.x * blockDim.x + threadIdx.x;
  if (idx >= Nout * Kp) return;
  int nout = idx / Kp, k = idx - nout * Kp;
  out[idx] = (k < K) ? f32_to_bf16(W[(size_t)k * Nout + nout]) : (unsigned short)0;
}

// ---------------- WMMA GEMM: C[M,Nout] = A[M,Kp] * Bt[Nout,Kp]^T (+bias) ----------------
// Block = 8 waves sharing one N-tile: the 16xKp B-panel is staged into LDS once per
// block (TDM async tensor load when available, cooperative copy otherwise), then each
// wave computes one 16x16 tile, B fed from ds_load_b128, A streamed from global.
__global__ void k_wmma_gemm(const unsigned short* __restrict__ A,
                            const unsigned short* __restrict__ Bt,
                            float* __restrict__ C, const float* __restrict__ bias,
                            int M, int Kp, int Nout) {
  __shared__ unsigned short Bs[16 * 512];   // 16 KB max (Kp <= 512)
  const int nt  = blockIdx.y;
  const int mt  = blockIdx.x * 8 + (threadIdx.x >> 5);
  const int lane = threadIdx.x & 31;
  const unsigned short* bsrc = Bt + (size_t)nt * 16 * Kp;

#if USE_TDM
  if ((threadIdx.x >> 5) == 0) {
    // NOTE: compile-only environment; flat->LDS offset truncation stands in for an
    // address-space-3 ptrtoint of Bs.
    tdm_load_tile_2d(bsrc, (unsigned)(size_t)(void*)&Bs[0], 16u, (unsigned)Kp, (unsigned)Kp);
    __builtin_amdgcn_s_wait_tensorcnt((short)0);
  }
#else
  for (int t = threadIdx.x; t < 2 * Kp; t += (int)blockDim.x)
    ((uint4*)Bs)[t] = ((const uint4*)bsrc)[t];
#endif
  __syncthreads();

  if (mt * 16 >= M) return;
  const int hs  = lane >> 4;       // half-wave select
  const int lid = lane & 15;

  // A fragment (16-bit A 16x32 layout): lanes 0-15 hold K {0..7,16..23}, lanes 16-31 {8..15,24..31}
  const unsigned short* arow = A + (size_t)(mt * 16 + lid) * Kp + hs * 8;
  // B fragment: lanes 0-15 hold K {0..7,16..23} of column N=lane, lanes 16-31 {8..15,24..31}
  const unsigned short* brow = Bs + lid * Kp + hs * 8;

  v8f acc = {};
  union Frag { bf16x16 v; uint4 q[2]; };
  for (int k = 0; k < Kp; k += 32) {
    Frag a, b;
    a.q[0] = *(const uint4*)(arow + k);
    a.q[1] = *(const uint4*)(arow + k + 16);
    b.q[0] = *(const uint4*)(brow + k);        // ds_load_b128
    b.q[1] = *(const uint4*)(brow + k + 16);
    __builtin_prefetch(arow + k + 32, 0, 0);   // global_prefetch_b8
    acc = __builtin_amdgcn_wmma_f32_16x16x32_bf16(
        false, a.v, false, b.v, (short)0, acc, false, false);
  }

  int row0 = mt * 16 + hs * 8;            // C layout: VGPR v -> M = v + hs*8
  int col  = nt * 16 + lid;               // N = lane (mod 16)
  float bv = bias ? bias[col] : 0.0f;
  #pragma unroll
  for (int v = 0; v < 8; ++v)
    C[(size_t)(row0 + v) * Nout + col] = acc[v] + bv;
}

// ---------------- aggregation (gcn message passing) ----------------

__global__ void k_self_bias(const float* __restrict__ h, const float* __restrict__ dinv,
                            const float* __restrict__ bias, float* __restrict__ out,
                            int n, int H) {
  int idx = blockIdx.x * blockDim.x + threadIdx.x;
  if (idx >= n * H) return;
  int i = idx / H, c = idx - i * H;
  float w = dinv[i];
  out[idx] = h[idx] * w * w + bias[c];
}

__global__ void k_scatter(const int* __restrict__ src, const int* __restrict__ dst,
                          const float* __restrict__ h, const float* __restrict__ dinv,
                          float* __restrict__ out, int e, int H) {
  int idx = blockIdx.x * blockDim.x + threadIdx.x;
  int c4n = H >> 2;
  if (idx >= e * c4n) return;
  int ed = idx / c4n;
  int c  = (idx - ed * c4n) << 2;
  int s = src[ed], d = dst[ed];
  float w = dinv[s] * dinv[d];
  const float4 hv = *(const float4*)(h + (size_t)s * H + c);
  float* o = out + (size_t)d * H + c;
  atomicAdd(o + 0, hv.x * w);
  atomicAdd(o + 1, hv.y * w);
  atomicAdd(o + 2, hv.z * w);
  atomicAdd(o + 3, hv.w * w);
}

// ---------------- batchnorm + leakyrelu ----------------

__global__ void k_bn_stats(const float* __restrict__ h, float* mean, float* istd, int n, int H) {
  int c = blockIdx.x;
  __shared__ float ss[256];
  __shared__ float ss2[256];
  float s = 0.0f, s2 = 0.0f;
  for (int i = threadIdx.x; i < n; i += 256) {
    float v = h[(size_t)i * H + c];
    s += v; s2 += v * v;
  }
  ss[threadIdx.x] = s; ss2[threadIdx.x] = s2;
  __syncthreads();
  for (int t = 128; t > 0; t >>= 1) {
    if ((int)threadIdx.x < t) {
      ss[threadIdx.x]  += ss[threadIdx.x + t];
      ss2[threadIdx.x] += ss2[threadIdx.x + t];
    }
    __syncthreads();
  }
  if (threadIdx.x == 0) {
    float m = ss[0] / (float)n;
    float var = ss2[0] / (float)n - m * m;   // biased, as torch BN training mode
    mean[c] = m;
    istd[c] = rsqrtf(var + EPSBN);
  }
}

__global__ void k_bn_leaky(const float* __restrict__ h, float* __restrict__ out,
                           float* __restrict__ skipdst,
                           const float* __restrict__ mean, const float* __restrict__ istd,
                           const float* __restrict__ g, const float* __restrict__ bt,
                           int n, int H) {
  int idx = blockIdx.x * blockDim.x + threadIdx.x;
  if (idx >= n * H) return;
  int c = idx % H;
  float v = g[c] * (h[idx] - mean[c]) * istd[c] + bt[c];
  v = (v >= 0.0f) ? v : SLOPE * v;
  out[idx] = v;
  if (skipdst) skipdst[idx] = v;
}

__global__ void k_concat(const float* __restrict__ a, const float* __restrict__ b,
                         float* __restrict__ out, int n, int d) {
  int idx = blockIdx.x * blockDim.x + threadIdx.x;
  int d2 = 2 * d;
  if (idx >= n * d2) return;
  int i = idx / d2, c = idx - i * d2;
  out[idx] = (c < d) ? a[(size_t)i * d + c] : b[(size_t)i * d + (c - d)];
}

__global__ void k_final(const float* __restrict__ x, const float* __restrict__ W,
                        const float* __restrict__ b, const float* __restrict__ xpos,
                        float* __restrict__ out, int n) {
  int i = blockIdx.x * blockDim.x + threadIdx.x;
  if (i >= n) return;
  float a0 = b[0], a1 = b[1], a2 = b[2];
  #pragma unroll
  for (int k = 0; k < 16; ++k) {
    float xv = x[(size_t)i * 16 + k];
    a0 += xv * W[k * 3 + 0];
    a1 += xv * W[k * 3 + 1];
    a2 += xv * W[k * 3 + 2];
  }
  out[i * 3 + 0] = xpos[i * 3 + 0] + a0;
  out[i * 3 + 1] = xpos[i * 3 + 1] + a1;
  out[i * 3 + 2] = xpos[i * 3 + 2] + a2;
}

// ---------------- host orchestration ----------------

static inline int cdiv(int a, int b) { return (a + b - 1) / b; }

extern "C" void kernel_launch(void* const* d_in, const int* in_sizes, int n_in,
                              void* d_out, int out_size, void* d_ws, size_t ws_size,
                              hipStream_t stream) {
  (void)in_sizes; (void)n_in; (void)out_size; (void)ws_size;
  static const int Hd[15] = {4,16,32,64,128,256,256,512,256,256,128,64,32,16,3};
  const int n = NNODES, e = NEDGES;

  const float* z1   = (const float*)d_in[0];
  const float* xpos = (const float*)d_in[1];
  const float* dm   = (const float*)d_in[2];
  const int*   ei   = (const int*)d_in[3];   // [2,E]: first E = src, next E = dst
  const float *Ws[13], *bs[13], *gm[13], *bt[13], *sW[6], *sb[6];
  for (int i = 0; i < 13; ++i) {
    Ws[i] = (const float*)d_in[4 + i];
    bs[i] = (const float*)d_in[17 + i];
    gm[i] = (const float*)d_in[30 + i];
    bt[i] = (const float*)d_in[43 + i];
  }
  for (int j = 0; j < 6; ++j) {
    sW[j] = (const float*)d_in[56 + j];
    sb[j] = (const float*)d_in[62 + j];
  }
  const float* linW = (const float*)d_in[68];
  const float* linb = (const float*)d_in[69];

  // carve workspace
  char* base = (char*)d_ws;
  size_t off = 0;
  auto alloc = [&](size_t bytes) -> char* {
    size_t a = (off + 255) & ~(size_t)255;
    off = a + bytes;
    return base + a;
  };
  float* P[3];
  for (int i = 0; i < 3; ++i) P[i] = (float*)alloc((size_t)n * 512 * sizeof(float));
  unsigned short* Abf  = (unsigned short*)alloc((size_t)n * 512 * sizeof(unsigned short));
  unsigned short* Wtbf = (unsigned short*)alloc((size_t)512 * 512 * sizeof(unsigned short));
  float* skipbuf = (float*)alloc((size_t)n * (16 + 32 + 64 + 128 + 256 + 256) * sizeof(float));
  float* deg  = (float*)alloc((size_t)n * sizeof(float));
  float* dinv = (float*)alloc((size_t)n * sizeof(float));
  float* mm   = (float*)alloc(16 * sizeof(float));
  float* mean = (float*)alloc(512 * sizeof(float));
  float* istd = (float*)alloc(512 * sizeof(float));

  float* skipP[6];
  { size_t o = 0; for (int i = 0; i < 6; ++i) { skipP[i] = skipbuf + o; o += (size_t)n * Hd[i + 1]; } }

  const int B = 256;
  const int mblk = cdiv(n / 16, 8);   // 8 M-tiles (waves) per block

  // degree + dinv (shared across layers)
  k_zero<<<cdiv(n, B), B, 0, stream>>>(deg, n);
  k_count_deg<<<cdiv(e, B), B, 0, stream>>>(ei + e, deg, e);
  k_dinv<<<cdiv(n, B), B, 0, stream>>>(deg, dinv, n);

  // input normalization
  k_minmax<<<1, 256, 0, stream>>>(z1, mm, n);
  k_build_input<<<cdiv(n, B), B, 0, stream>>>(z1, dm, mm, P[0], n);

  float* xcur = P[0];
  auto freebuf = [&](float* a, float* b) -> float* {
    for (int i = 0; i < 3; ++i) if (P[i] != a && P[i] != b) return P[i];
    return P[0];
  };

  for (int i = 0; i < 13; ++i) {
    int Hin = Hd[i];
    if (i >= 8) {  // decoder skip-concat linear
      int j = 13 - i;
      int d = Hd[j + 1];  // == Hin
      float* cat = freebuf(xcur, nullptr);
      k_concat<<<cdiv(n * 2 * d, B), B, 0, stream>>>(skipP[j], xcur, cat, n, d);
      int K = 2 * d, Kp = K;  // 2d is always a multiple of 32 here
      k_cvt_bf16<<<cdiv(n * Kp, B), B, 0, stream>>>(cat, Abf, n, K, Kp);
      k_cvt_wt<<<cdiv(d * Kp, B), B, 0, stream>>>(sW[j], Wtbf, K, d, Kp);
      float* gout = freebuf(xcur, cat);
      k_wmma_gemm<<<dim3(mblk, d / 16), 256, 0, stream>>>(Abf, Wtbf, gout, sb[j], n, Kp, d);
      xcur = gout;
    }
    int Hout = Hd[i + 1];
    int Kp = (Hin + 31) & ~31;
    k_cvt_bf16<<<cdiv(n * Kp, B), B, 0, stream>>>(xcur, Abf, n, Hin, Kp);
    k_cvt_wt<<<cdiv(Hout * Kp, B), B, 0, stream>>>(Ws[i], Wtbf, Hin, Hout, Kp);
    float* hbuf = freebuf(xcur, nullptr);
    k_wmma_gemm<<<dim3(mblk, Hout / 16), 256, 0, stream>>>(Abf, Wtbf, hbuf, nullptr, n, Kp, Hout);
    float* agg = freebuf(xcur, hbuf);
    k_self_bias<<<cdiv(n * Hout, B), B, 0, stream>>>(hbuf, dinv, bs[i], agg, n, Hout);
    k_scatter<<<cdiv(e * (Hout / 4), B), B, 0, stream>>>(ei, ei + e, hbuf, dinv, agg, e, Hout);
    k_bn_stats<<<Hout, 256, 0, stream>>>(agg, mean, istd, n, Hout);
    float* xout = hbuf;  // hbuf free after scatter
    k_bn_leaky<<<cdiv(n * Hout, B), B, 0, stream>>>(agg, xout, (i <= 5) ? skipP[i] : nullptr,
                                                    mean, istd, gm[i], bt[i], n, Hout);
    xcur = xout;
  }

  k_final<<<cdiv(n, B), B, 0, stream>>>(xcur, linW, linb, xpos, (float*)d_out, n);
}
